// CVKANTimeSeries_1477468750539
// MI455X (gfx1250) — matrix operand order repaired and checked by reading, not compiled, not run on hardware
//
#include <hip/hip_runtime.h>
#include <math.h>

#define BATCH  8
#define SEQ    1024
#define DMODEL 256
#define HID    32
#define INDIM  64
#define NLAYER 4
#define MTOT   (BATCH * SEQ)   // 8192

#define XPITCH 68    // 64 + 4 pad: A-fragment rows hit distinct LDS banks; 16B aligned
#define WPITCH 132   // 128 + 4 pad: B-fragment column reads conflict-free; 16B aligned

typedef _Float16 v16h __attribute__((ext_vector_type(16)));
typedef float    v8f  __attribute__((ext_vector_type(8)));
typedef int      ai4  __attribute__((vector_size(4 * sizeof(int))));   // matches builtin proto

__device__ __forceinline__ float gelu_exact(float x) {
    // torch nn.GELU default: 0.5*x*(1+erf(x/sqrt(2)))
    return 0.5f * x * (1.0f + erff(x * 0.70710678118654752f));
}

// Stage one 16-byte segment global -> LDS, preferring the CDNA5 async path
// (GLOBAL_LOAD_ASYNC_TO_LDS_B128, tracked by ASYNCcnt).
__device__ __forceinline__ void stage16(const float* g, float* l) {
#if defined(__has_builtin) && __has_builtin(__builtin_amdgcn_global_load_async_to_lds_b128)
    __builtin_amdgcn_global_load_async_to_lds_b128(
        (__attribute__((address_space(1))) ai4*)g,
        (__attribute__((address_space(3))) ai4*)l, 0, 0);
#else
    *(float4*)l = *(const float4*)g;
#endif
}

// ---------------------------------------------------------------------------
// Kernel 1: zr/zi = x @ {er_w,ei_w} + bias, then complex positional rotation.
// Block = 8 waves covering one 16-row M tile x 128 columns. x tile and both
// weight tiles are staged to LDS once per block (async-to-LDS), fragments are
// built from LDS (bank-conflict-free via padded pitch), 2 WMMA K-steps per
// accumulator. Epilogue writes TRANSPOSED [B,D,S] so the scan is coalesced.
// ---------------------------------------------------------------------------
__global__ __launch_bounds__(256)
void embed_rotate_wmma(const float* __restrict__ x,
                       const float* __restrict__ er_w, const float* __restrict__ er_b,
                       const float* __restrict__ ei_w, const float* __restrict__ ei_b,
                       float* __restrict__ wsR, float* __restrict__ wsI)
{
    __shared__ float sX [16 * XPITCH];   // 16 x 64 f32 x-tile   (~4.3 KB)
    __shared__ float sWr[64 * WPITCH];   // 64 x 128 f32 er_w    (~33.8 KB)
    __shared__ float sWi[64 * WPITCH];   // 64 x 128 f32 ei_w    (~33.8 KB)

    const int tid   = threadIdx.x;
    const int lane  = tid & 31;
    const int wave  = tid >> 5;
    const int m0    = (blockIdx.x >> 1) * 16;   // 512 M tiles
    const int ncol0 = (blockIdx.x & 1) * 128;   // two column halves of D=256

    // ---- stage x tile: 1024 floats, one b128 segment per thread ----
    {
        int row = tid >> 4, seg = tid & 15;
        stage16(x + (m0 + row) * INDIM + seg * 4, sX + row * XPITCH + seg * 4);
    }
    // ---- stage er_w/ei_w tiles: 64x128 floats each (32 segs/row) ----
    #pragma unroll
    for (int it = 0; it < 8; ++it) {
        int idx = it * 256 + tid;            // 2048 segments
        int row = idx >> 5, seg = idx & 31;
        int gofs = row * DMODEL + ncol0 + seg * 4;
        int lofs = row * WPITCH + seg * 4;
        stage16(er_w + gofs, sWr + lofs);
        stage16(ei_w + gofs, sWi + lofs);
    }
    asm volatile("s_wait_asynccnt 0x0" ::: "memory");
    __syncthreads();

    const int hi = (lane >> 4) & 1;
    const int r  = lane & 15;                 // A row within tile
    const int c  = wave * 16 + (lane & 15);   // B column within 128-col stripe

    v8f cr = {};
    v8f ci = {};
    #pragma unroll
    for (int kc = 0; kc < 2; ++kc) {
        const int kbase = kc * 32;
        v16h a, br, bi;
        #pragma unroll
        for (int e = 0; e < 8; ++e) {         // A: 16-bit 16x32 layout (ISA 7.12.2)
            a[e]     = (_Float16)sX[r * XPITCH + kbase + hi * 8 + e];
            a[e + 8] = (_Float16)sX[r * XPITCH + kbase + 16 + hi * 8 + e];
        }
        #pragma unroll
        for (int e = 0; e < 16; ++e) {        // B: 16-bit 32x16 layout, same K order as A
            int k = kbase + (e < 8 ? e : e + 8) + hi * 8;
            br[e] = (_Float16)sWr[k * WPITCH + c];
            bi[e] = (_Float16)sWi[k * WPITCH + c];
        }
        cr = __builtin_amdgcn_wmma_f32_16x16x32_f16(false, a, false, br, (short)0, cr, false, false);
        ci = __builtin_amdgcn_wmma_f32_16x16x32_f16(false, a, false, bi, (short)0, ci, false, false);
    }

    const int   n    = ncol0 + c;
    const float freq = __expf(-logf(10000.0f) * (float)n / (float)DMODEL);
    const float brb  = er_b[n];
    const float bib  = ei_b[n];
    #pragma unroll
    for (int j = 0; j < 8; ++j) {
        int   m  = m0 + j + 8 * hi;        // C/D layout: VGPR j holds rows j and j+8
        int   s  = m & (SEQ - 1);
        int   bb = m >> 10;
        float zr = cr[j] + brb;
        float zi = ci[j] + bib;
        float th = (float)s * freq;
        float sn, cs;
        __sincosf(th, &sn, &cs);
        float orr = zr * cs - zi * sn;
        float oii = zr * sn + zi * cs;
        int addr = (bb * DMODEL + n) * SEQ + s;   // transposed [B,D,S]
        wsR[addr] = orr;
        wsI[addr] = oii;
    }
}

// ---------------------------------------------------------------------------
// Kernel 2: all 4 polarizing layers fused. One block per (b,d) chain (2048
// blocks x 256 threads; 4 elements/thread, float4-coalesced). Causal cumsum =
// local scan + wave shuffle scan + LDS cross-wave offsets. Tiny MLPs applied
// per element with weights broadcast from LDS. zr/zi live in registers for
// the whole 4-layer loop -> intermediate touched once.
// ---------------------------------------------------------------------------
__global__ __launch_bounds__(256)
void polar_layers(float* __restrict__ wsR, float* __restrict__ wsI,
                  const float* __restrict__ pm_w1, const float* __restrict__ pm_b1,
                  const float* __restrict__ pm_w2, const float* __restrict__ pm_b2,
                  const float* __restrict__ pp_w1, const float* __restrict__ pp_b1,
                  const float* __restrict__ pp_w2, const float* __restrict__ pp_b2,
                  const float* __restrict__ mag_scale)
{
    __shared__ float sWm1[NLAYER][HID], sBm1[NLAYER][HID], sWm2[NLAYER][HID];
    __shared__ float sWp1[NLAYER][2 * HID], sBp1[NLAYER][HID], sWp2[NLAYER][2 * HID];
    __shared__ float sBm2[NLAYER], sBp2[NLAYER][2], sMsc[NLAYER];
    __shared__ float partR[8], partI[8];

    const int tid = threadIdx.x;
    for (int i = tid; i < NLAYER * HID; i += 256) {
        sWm1[i >> 5][i & 31] = pm_w1[i];
        sBm1[i >> 5][i & 31] = pm_b1[i];
        sWm2[i >> 5][i & 31] = pm_w2[i];
        sBp1[i >> 5][i & 31] = pp_b1[i];
    }
    for (int i = tid; i < NLAYER * 2 * HID; i += 256) {
        sWp1[i / (2 * HID)][i % (2 * HID)] = pp_w1[i];
        sWp2[i / (2 * HID)][i % (2 * HID)] = pp_w2[i];
    }
    if (tid < NLAYER)     { sBm2[tid] = pm_b2[tid]; sMsc[tid] = mag_scale[tid]; }
    if (tid < 2 * NLAYER) { sBp2[tid >> 1][tid & 1] = pp_b2[tid]; }
    __syncthreads();

    const int    lane = tid & 31;
    const int    wv   = tid >> 5;
    const size_t base = (size_t)blockIdx.x * SEQ + (size_t)tid * 4;
    float4 r4 = *(const float4*)(wsR + base);
    float4 i4 = *(const float4*)(wsI + base);
    float zr[4] = {r4.x, r4.y, r4.z, r4.w};
    float zi[4] = {i4.x, i4.y, i4.z, i4.w};

    for (int l = 0; l < NLAYER; ++l) {
        // --- causal cumulative sums ---
        float cr[4], cic[4];
        cr[0] = zr[0]; cic[0] = zi[0];
        #pragma unroll
        for (int q = 1; q < 4; ++q) { cr[q] = cr[q - 1] + zr[q]; cic[q] = cic[q - 1] + zi[q]; }
        float tr = cr[3], ti = cic[3];
        float ir = tr, ii = ti;
        #pragma unroll
        for (int d = 1; d < 32; d <<= 1) {
            float ur = __shfl_up(ir, d);
            float ui = __shfl_up(ii, d);
            if (lane >= d) { ir += ur; ii += ui; }
        }
        if (lane == 31) { partR[wv] = ir; partI[wv] = ii; }
        __syncthreads();
        float offR = 0.f, offI = 0.f;
        for (int p = 0; p < wv; ++p) { offR += partR[p]; offI += partI[p]; }
        const float exR = offR + (ir - tr);   // exclusive prefix before this thread
        const float exI = offI + (ii - ti);

        // --- per-element polar features ---
        float lm[4], pr_[4], pi_[4], macc[4], phr[4], phi_[4];
        #pragma unroll
        for (int q = 0; q < 4; ++q) {
            float cnt = (float)(tid * 4 + q + 1);
            float Ar  = (exR + cr[q])  / cnt;
            float Ai  = (exI + cic[q]) / cnt;
            float mag = sqrtf(Ar * Ar + Ai * Ai);
            lm[q]  = logf(mag + 1e-6f);
            float inv = 1.0f / (mag + 1e-6f);
            pr_[q] = Ar * inv;
            pi_[q] = Ai * inv;
            macc[q] = 0.f; phr[q] = 0.f; phi_[q] = 0.f;
        }

        // --- psi_mag (1->32->1) and psi_phase (2->32->2), fused over H ---
        const float* Wm1 = sWm1[l]; const float* Bm1 = sBm1[l]; const float* Wm2 = sWm2[l];
        const float* Wp1 = sWp1[l]; const float* Bp1 = sBp1[l]; const float* Wp2 = sWp2[l];
        #pragma unroll 4
        for (int h = 0; h < HID; ++h) {
            float w1m = Wm1[h], b1m = Bm1[h], w2m = Wm2[h];
            float wp0 = Wp1[h], wp1v = Wp1[HID + h], bpv = Bp1[h];
            float wq0 = Wp2[2 * h], wq1 = Wp2[2 * h + 1];
            #pragma unroll
            for (int q = 0; q < 4; ++q) {
                float g1 = gelu_exact(fmaf(lm[q], w1m, b1m));
                macc[q] = fmaf(g1, w2m, macc[q]);
                float g2 = gelu_exact(fmaf(pr_[q], wp0, fmaf(pi_[q], wp1v, bpv)));
                phr[q]  = fmaf(g2, wq0, phr[q]);
                phi_[q] = fmaf(g2, wq1, phi_[q]);
            }
        }

        float bm2 = sBm2[l], msc = sMsc[l], bp0 = sBp2[l][0], bp1 = sBp2[l][1];
        #pragma unroll
        for (int q = 0; q < 4; ++q) {
            float lo  = lm[q] + msc * (macc[q] + bm2);
            float vr  = phr[q] + bp0;
            float vi  = phi_[q] + bp1;
            float nrm = fmaxf(sqrtf(vr * vr + vi * vi), 1e-12f);
            float rsc = expf(lo) / nrm;
            zr[q] = fmaf(rsc, vr, zr[q]);
            zi[q] = fmaf(rsc, vi, zi[q]);
        }
        __syncthreads();   // partR/partI reused next layer
    }

    *(float4*)(wsR + base) = make_float4(zr[0], zr[1], zr[2], zr[3]);
}

// ---------------------------------------------------------------------------
// Kernel 3: decode. feats(=zr, [B,D,S] layout) @ op_w1 (K=256 -> N=32) via
// WMMA (2 N-tiles, 8 K-steps), then GELU + 32->1 dot via shuffle reduction.
// ---------------------------------------------------------------------------
__global__ __launch_bounds__(256)
void decode_wmma(const float* __restrict__ wsR,
                 const float* __restrict__ op_w1, const float* __restrict__ op_b1,
                 const float* __restrict__ op_w2, const float* __restrict__ op_b2,
                 float* __restrict__ out)
{
    const int lane = threadIdx.x & 31;
    const int wave = threadIdx.x >> 5;
    const int gw   = blockIdx.x * 8 + wave;   // 512 waves, one 16-row tile each
    const int m0   = gw * 16;
    const int hi   = (lane >> 4) & 1;
    const int arow = m0 + (lane & 15);
    const int bb   = m0 >> 10;                // 16 | 1024 -> whole tile in one batch
    const int srow = arow & (SEQ - 1);

    v8f c0 = {};
    v8f c1 = {};
    for (int kc = 0; kc < 8; ++kc) {
        const int kbase = kc * 32;
        v16h a, b0, b1v;
        #pragma unroll
        for (int e = 0; e < 16; ++e) {
            int ka = kbase + (e < 8 ? e : e + 8) + hi * 8;
            a[e] = (_Float16)wsR[(size_t)(bb * DMODEL + ka) * SEQ + srow];
            int kb = kbase + (e < 8 ? e : e + 8) + hi * 8;
            int nn = lane & 15;
            b0[e]  = (_Float16)op_w1[kb * HID + nn];
            b1v[e] = (_Float16)op_w1[kb * HID + nn + 16];
        }
        c0 = __builtin_amdgcn_wmma_f32_16x16x32_f16(false, a, false, b0,  (short)0, c0, false, false);
        c1 = __builtin_amdgcn_wmma_f32_16x16x32_f16(false, a, false, b1v, (short)0, c1, false, false);
    }

    const int   n    = lane & 15;
    const float ob1a = op_b1[n],      ob1b = op_b1[n + 16];
    const float ow2a = op_w2[n],      ow2b = op_w2[n + 16];
    const float ob2  = op_b2[0];
    #pragma unroll
    for (int j = 0; j < 8; ++j) {
        float v = gelu_exact(c0[j] + ob1a) * ow2a + gelu_exact(c1[j] + ob1b) * ow2b;
        #pragma unroll
        for (int d = 1; d < 16; d <<= 1) v += __shfl_xor(v, d);  // reduce within 16-lane halves
        if (n == 0) out[m0 + j + 8 * hi] = v + ob2;
    }
}

// ---------------------------------------------------------------------------
extern "C" void kernel_launch(void* const* d_in, const int* in_sizes, int n_in,
                              void* d_out, int out_size, void* d_ws, size_t ws_size,
                              hipStream_t stream)
{
    (void)in_sizes; (void)n_in; (void)out_size; (void)ws_size;

    const float* x     = (const float*)d_in[0];
    const float* er_w  = (const float*)d_in[1];
    const float* er_b  = (const float*)d_in[2];
    const float* ei_w  = (const float*)d_in[3];
    const float* ei_b  = (const float*)d_in[4];
    const float* pm_w1 = (const float*)d_in[5];
    const float* pm_b1 = (const float*)d_in[6];
    const float* pm_w2 = (const float*)d_in[7];
    const float* pm_b2 = (const float*)d_in[8];
    const float* pp_w1 = (const float*)d_in[9];
    const float* pp_b1 = (const float*)d_in[10];
    const float* pp_w2 = (const float*)d_in[11];
    const float* pp_b2 = (const float*)d_in[12];
    const float* msc   = (const float*)d_in[13];
    const float* op_w1 = (const float*)d_in[14];
    const float* op_b1 = (const float*)d_in[15];
    const float* op_w2 = (const float*)d_in[16];
    const float* op_b2 = (const float*)d_in[17];
    float*       out   = (float*)d_out;

    float* wsR = (float*)d_ws;                               // [B,D,S] = 8 MB
    float* wsI = wsR + (size_t)BATCH * SEQ * DMODEL;         // [B,D,S] = 8 MB

    // 512 M tiles x 2 column halves, 8 waves/block (one 16x16 N-tile each)
    embed_rotate_wmma<<<1024, 256, 0, stream>>>(x, er_w, er_b, ei_w, ei_b, wsR, wsI);
    // one block per (b,d) chain
    polar_layers<<<BATCH * DMODEL, 256, 0, stream>>>(wsR, wsI,
                                                     pm_w1, pm_b1, pm_w2, pm_b2,
                                                     pp_w1, pp_b1, pp_w2, pp_b2, msc);
    // 512 M-tiles, 8 waves/block
    decode_wmma<<<64, 256, 0, stream>>>(wsR, op_w1, op_b1, op_w2, op_b2, out);
}
